// MambaBlock_29472065585581
// MI455X (gfx1250) — matrix operand order repaired and checked by reading, not compiled, not run on hardware
//
#include <hip/hip_runtime.h>
#include <hip/hip_bf16.h>
#include <math.h>

// ---------------------------------------------------------------------------
// Mamba block forward for MI455X (gfx1250, wave32).
//
//   1. f32->bf16 conversions of x and all weight matrices (workspace copies)
//   2. xz   = x @ W_in^T            (WMMA bf16 GEMM, f32 accum, 2048x8192x2048)
//   3. xc   = silu(causal_conv4(xi)), also cast to bf16
//   4. dbc  = xc @ W_x^T            (WMMA GEMM, N=160)
//   5. dtlo = bf16(dbc[:, :128])
//   6. delta= softplus(dtlo @ W_dt^T + b_dt)   (WMMA GEMM + fused epilogue)
//   7. chunked selective scan (3 passes) -> ys
//   8. y_bf = bf16((ys + xc*D) * silu(z))
//   9. out  = y_bf @ W_out^T        (WMMA GEMM, f32 out)
//
// All global traffic is cast to address_space(1) so the backend emits
// global_load/global_store (pure LOADcnt/STOREcnt) instead of FLAT ops
// that also tie up DScnt and the LDS arbitration path.
// ---------------------------------------------------------------------------

#define L_SEQ   2048
#define DIM     2048
#define INNER   4096
#define STATE   16
#define KCONV   4
#define DT_RANK 128
#define NPROJ   (DT_RANK + 2 * STATE)   // 160
#define NCHUNK  32
#define CHUNK   (L_SEQ / NCHUNK)        // 64

typedef __attribute__((ext_vector_type(16))) __bf16       v16bf;
typedef __attribute__((ext_vector_type(8)))  float        v8f;
typedef __attribute__((ext_vector_type(4)))  unsigned int u32x4;

// Global-address-space access helpers (force global_load_*/global_store_*)
#define GLD_U4(p)    (*(const __attribute__((address_space(1))) u32x4*)(p))
#define GLD_F(p)     (*(const __attribute__((address_space(1))) float*)(p))
#define GST_F(p, v)  (*(__attribute__((address_space(1))) float*)(p) = (v))
#define GST_BF(p, v) (*(__attribute__((address_space(1))) __bf16*)(p) = (v))

union FragAB {
    v16bf v;
    u32x4 q[2];
};

__device__ __forceinline__ float silu_f(float v) {
    return v / (1.0f + __expf(-v));
}

// ---------------------------------------------------------------------------
// f32 -> bf16 conversion
// ---------------------------------------------------------------------------
__global__ void cvt_f32_bf16_kernel(const float* __restrict__ src,
                                    __bf16* __restrict__ dst, int n) {
    int i = blockIdx.x * blockDim.x + threadIdx.x;
    if (i < n) GST_BF(dst + i, (__bf16)GLD_F(src + i));
}

// Slice dbc[:, :DT_RANK] -> bf16
__global__ void cvt_slice_bf16_kernel(const float* __restrict__ src,
                                      __bf16* __restrict__ dst,
                                      int rows, int src_ld, int cols) {
    int i = blockIdx.x * blockDim.x + threadIdx.x;
    if (i >= rows * cols) return;
    int r = i / cols, c = i - r * cols;
    GST_BF(dst + i, (__bf16)GLD_F(src + (size_t)r * src_ld + c));
}

// ---------------------------------------------------------------------------
// Generic bf16 WMMA GEMM: C[M,N] = A[M,K] @ BT[N,K]^T   (f32 accumulate)
// Block: 256 threads = 8 waves; block tile 128x256; wave tile 64x64 (4x4).
// 16 b128 loads per 16 WMMAs; each fragment reused 4x.
// epi==1: C = softplus(C + bias[col])
// ---------------------------------------------------------------------------
__global__ __launch_bounds__(256)
void gemm_bf16_wmma_kernel(const __bf16* __restrict__ A,
                           const __bf16* __restrict__ BT,
                           float* __restrict__ C,
                           int M, int N, int K,
                           const float* __restrict__ bias, int epi) {
    const int wid  = threadIdx.x >> 5;    // 0..7
    const int lane = threadIdx.x & 31;
    const int l16  = lane & 15;
    const int lhi  = lane >> 4;           // 0 or 1

    const int mBase = blockIdx.x * 128 + (wid & 1) * 64;   // 2 waves along M
    const int nBase = blockIdx.y * 256 + (wid >> 1) * 64;  // 4 waves along N

    v8f acc[4][4];
#pragma unroll
    for (int mi = 0; mi < 4; ++mi)
#pragma unroll
        for (int ni = 0; ni < 4; ++ni) acc[mi][ni] = (v8f){};

    // A-fragment per-lane source rows (16-bit A layout, 16x32):
    //   lane<16 : M=l16, K chunks [0..7],[16..23]; lane>=16: K [8..15],[24..31]
    const __bf16* pA[4];
#pragma unroll
    for (int mi = 0; mi < 4; ++mi) {
        int r = mBase + mi * 16 + l16;
        r = r < M ? r : M - 1;
        pA[mi] = A + (size_t)r * K + (lhi << 3);
    }
    // B-fragment per-lane source rows (B 32x16): lane holds column N=l16,
    //   lane<16: K [0..7],[16..23]; lane>=16: K [8..15],[24..31].
    const __bf16* pB[4];
#pragma unroll
    for (int ni = 0; ni < 4; ++ni) {
        int r = nBase + ni * 16 + l16;
        r = r < N ? r : N - 1;
        pB[ni] = BT + (size_t)r * K + (lhi << 3);
    }

    for (int k0 = 0; k0 < K; k0 += 32) {
        FragAB a[4], b[4];
#pragma unroll
        for (int mi = 0; mi < 4; ++mi) {
            a[mi].q[0] = GLD_U4(pA[mi]);
            a[mi].q[1] = GLD_U4(pA[mi] + 16);
            __builtin_prefetch(pA[mi] + 256, 0, 1);   // 8 K-steps ahead
            pA[mi] += 32;
        }
#pragma unroll
        for (int ni = 0; ni < 4; ++ni) {
            b[ni].q[0] = GLD_U4(pB[ni]);
            b[ni].q[1] = GLD_U4(pB[ni] + 16);
            __builtin_prefetch(pB[ni] + 256, 0, 1);
            pB[ni] += 32;
        }
#pragma unroll
        for (int mi = 0; mi < 4; ++mi)
#pragma unroll
            for (int ni = 0; ni < 4; ++ni)
                acc[mi][ni] = __builtin_amdgcn_wmma_f32_16x16x32_bf16(
                    false, a[mi].v, false, b[ni].v,
                    (short)0, acc[mi][ni], false, false);
    }

    // C/D layout: VGPR r -> row (r + 8*lhi), col l16.
    const int rowOff = lhi << 3;
#pragma unroll
    for (int mi = 0; mi < 4; ++mi) {
#pragma unroll
        for (int ni = 0; ni < 4; ++ni) {
#pragma unroll
            for (int r = 0; r < 8; ++r) {
                int row = mBase + mi * 16 + rowOff + r;
                int col = nBase + ni * 16 + l16;
                if (row < M && col < N) {
                    float v = acc[mi][ni][r];
                    if (epi == 1) {
                        v += GLD_F(bias + col);
                        v = (v > 20.0f) ? v : log1pf(__expf(v));
                    }
                    GST_F(C + (size_t)row * N + col, v);
                }
            }
        }
    }
}

// ---------------------------------------------------------------------------
// Causal depthwise conv (K=4) + bias + SiLU; writes f32 and bf16 copies.
// xi = xz[:, :INNER]  (row stride 2*INNER)
// ---------------------------------------------------------------------------
__global__ void conv_silu_kernel(const float* __restrict__ xz,
                                 const float* __restrict__ conv_w,
                                 const float* __restrict__ conv_b,
                                 float* __restrict__ xc,
                                 __bf16* __restrict__ xc_bf) {
    int idx = blockIdx.x * blockDim.x + threadIdx.x;
    if (idx >= L_SEQ * INNER) return;
    int t = idx / INNER, d = idx - t * INNER;
    float acc = GLD_F(conv_b + d);
#pragma unroll
    for (int j = 0; j < KCONV; ++j) {
        int tt = t - (KCONV - 1) + j;
        if (tt >= 0)
            acc += GLD_F(xz + (size_t)tt * (2 * INNER) + d) * GLD_F(conv_w + d * KCONV + j);
    }
    float v = silu_f(acc);
    GST_F(xc + idx, v);
    GST_BF(xc_bf + idx, (__bf16)v);
}

// ---------------------------------------------------------------------------
// Selective scan, pass A: per-chunk local scan with h0=0.
// Also records per-chunk decay product P = prod(dA) and final local h.
// ---------------------------------------------------------------------------
__global__ void scan_chunk_kernel(const float* __restrict__ delta,
                                  const float* __restrict__ xc,
                                  const float* __restrict__ dbc,
                                  const float* __restrict__ A_log,
                                  float* __restrict__ ys,
                                  float* __restrict__ hLast,
                                  float* __restrict__ prodOut) {
    int d = blockIdx.x * blockDim.x + threadIdx.x;  // 0..INNER-1
    int c = blockIdx.y;                             // chunk
    float Ac[STATE], h[STATE], P[STATE];
#pragma unroll
    for (int n = 0; n < STATE; ++n) {
        Ac[n] = -__expf(GLD_F(A_log + (size_t)d * STATE + n));
        h[n] = 0.0f;
        P[n] = 1.0f;
    }
    int t0 = c * CHUNK;
    for (int t = t0; t < t0 + CHUNK; ++t) {
        float dt = GLD_F(delta + (size_t)t * INNER + d);
        float dx = dt * GLD_F(xc + (size_t)t * INNER + d);
        const float* Bv = dbc + (size_t)t * NPROJ + DT_RANK;
        const float* Cv = Bv + STATE;
        float y = 0.0f;
#pragma unroll
        for (int n = 0; n < STATE; ++n) {
            float dA = __expf(dt * Ac[n]);
            h[n] = dA * h[n] + dx * GLD_F(Bv + n);
            P[n] *= dA;
            y += h[n] * GLD_F(Cv + n);
        }
        GST_F(ys + (size_t)t * INNER + d, y);
    }
    size_t o = ((size_t)c * INNER + d) * STATE;
#pragma unroll
    for (int n = 0; n < STATE; ++n) {
        GST_F(hLast + o + n, h[n]);
        GST_F(prodOut + o + n, P[n]);
    }
}

// Pass B: sequential carry over 32 chunks; one thread per (d, n).
__global__ void scan_carry_kernel(const float* __restrict__ hLast,
                                  const float* __restrict__ prod,
                                  float* __restrict__ carryIn) {
    int i = blockIdx.x * blockDim.x + threadIdx.x;
    if (i >= INNER * STATE) return;
    float carry = 0.0f;
    for (int c = 0; c < NCHUNK; ++c) {
        size_t o = (size_t)c * INNER * STATE + i;
        GST_F(carryIn + o, carry);
        carry = GLD_F(prod + o) * carry + GLD_F(hLast + o);
    }
}

// Pass C: fixup — ys[t] += sum_n (prod_{s<=t} dA_s) * carry[n] * C_t[n]
__global__ void scan_fixup_kernel(const float* __restrict__ delta,
                                  const float* __restrict__ dbc,
                                  const float* __restrict__ A_log,
                                  const float* __restrict__ carryIn,
                                  float* __restrict__ ys) {
    int d = blockIdx.x * blockDim.x + threadIdx.x;
    int c = blockIdx.y;
    if (c == 0) return;
    float Ac[STATE], cp[STATE];
    size_t o = ((size_t)c * INNER + d) * STATE;
#pragma unroll
    for (int n = 0; n < STATE; ++n) {
        Ac[n] = -__expf(GLD_F(A_log + (size_t)d * STATE + n));
        cp[n] = GLD_F(carryIn + o + n);
    }
    int t0 = c * CHUNK;
    for (int t = t0; t < t0 + CHUNK; ++t) {
        float dt = GLD_F(delta + (size_t)t * INNER + d);
        const float* Cv = dbc + (size_t)t * NPROJ + DT_RANK + STATE;
        float ycorr = 0.0f;
#pragma unroll
        for (int n = 0; n < STATE; ++n) {
            cp[n] *= __expf(dt * Ac[n]);
            ycorr += cp[n] * GLD_F(Cv + n);
        }
        float cur = GLD_F(ys + (size_t)t * INNER + d);
        GST_F(ys + (size_t)t * INNER + d, cur + ycorr);
    }
}

// ---------------------------------------------------------------------------
// Gate: y_bf = bf16((ys + xc*D) * silu(z)),  z = xz[:, INNER:]
// ---------------------------------------------------------------------------
__global__ void gate_kernel(const float* __restrict__ ys,
                            const float* __restrict__ xc,
                            const float* __restrict__ D,
                            const float* __restrict__ xz,
                            __bf16* __restrict__ y_bf) {
    int idx = blockIdx.x * blockDim.x + threadIdx.x;
    if (idx >= L_SEQ * INNER) return;
    int t = idx / INNER, d = idx - t * INNER;
    float zv = GLD_F(xz + (size_t)t * (2 * INNER) + INNER + d);
    float v = (GLD_F(ys + idx) + GLD_F(xc + idx) * GLD_F(D + d)) * silu_f(zv);
    GST_BF(y_bf + idx, (__bf16)v);
}

// ---------------------------------------------------------------------------
// Host launcher
// ---------------------------------------------------------------------------
extern "C" void kernel_launch(void* const* d_in, const int* in_sizes, int n_in,
                              void* d_out, int out_size, void* d_ws, size_t ws_size,
                              hipStream_t stream) {
    const float* x      = (const float*)d_in[0];  // (L, DIM)
    const float* W_in   = (const float*)d_in[1];  // (2*INNER, DIM)
    const float* conv_w = (const float*)d_in[2];  // (INNER, 4)
    const float* conv_b = (const float*)d_in[3];  // (INNER,)
    const float* W_x    = (const float*)d_in[4];  // (NPROJ, INNER)
    const float* W_dt   = (const float*)d_in[5];  // (INNER, DT_RANK)
    const float* b_dt   = (const float*)d_in[6];  // (INNER,)
    const float* A_log  = (const float*)d_in[7];  // (INNER, STATE)
    const float* Dvec   = (const float*)d_in[8];  // (INNER,)
    const float* W_out  = (const float*)d_in[9];  // (DIM, INNER)
    float* out = (float*)d_out;                   // (L, DIM)

    // Workspace carve-up (256B aligned)
    char* ws = (char*)d_ws;
    size_t off = 0;
    auto alloc = [&](size_t bytes) -> char* {
        char* p = ws + off;
        off = (off + bytes + 255) & ~(size_t)255;
        return p;
    };
    __bf16* x_bf    = (__bf16*)alloc((size_t)L_SEQ * DIM * 2);
    __bf16* Win_bf  = (__bf16*)alloc((size_t)2 * INNER * DIM * 2);
    float*  xz      = (float*) alloc((size_t)L_SEQ * 2 * INNER * 4);
    float*  xc      = (float*) alloc((size_t)L_SEQ * INNER * 4);
    __bf16* xc_bf   = (__bf16*)alloc((size_t)L_SEQ * INNER * 2);
    __bf16* Wx_bf   = (__bf16*)alloc((size_t)NPROJ * INNER * 2);
    float*  dbc     = (float*) alloc((size_t)L_SEQ * NPROJ * 4);
    __bf16* dtlo_bf = (__bf16*)alloc((size_t)L_SEQ * DT_RANK * 2);
    __bf16* Wdt_bf  = (__bf16*)alloc((size_t)INNER * DT_RANK * 2);
    float*  delta   = (float*) alloc((size_t)L_SEQ * INNER * 4);
    float*  ysbuf   = (float*) alloc((size_t)L_SEQ * INNER * 4);
    float*  hLast   = (float*) alloc((size_t)NCHUNK * INNER * STATE * 4);
    float*  prodP   = (float*) alloc((size_t)NCHUNK * INNER * STATE * 4);
    float*  carry   = (float*) alloc((size_t)NCHUNK * INNER * STATE * 4);
    __bf16* y_bf    = (__bf16*)alloc((size_t)L_SEQ * INNER * 2);
    __bf16* Wout_bf = (__bf16*)alloc((size_t)DIM * INNER * 2);
    (void)ws_size; (void)n_in; (void)in_sizes; (void)out_size;

    const int T = 256;
    auto blocks = [](long n, int t) { return (int)((n + t - 1) / t); };

    // 1. bf16 conversions
    cvt_f32_bf16_kernel<<<blocks((long)L_SEQ * DIM, T), T, 0, stream>>>(x, x_bf, L_SEQ * DIM);
    cvt_f32_bf16_kernel<<<blocks((long)2 * INNER * DIM, T), T, 0, stream>>>(W_in, Win_bf, 2 * INNER * DIM);
    cvt_f32_bf16_kernel<<<blocks((long)NPROJ * INNER, T), T, 0, stream>>>(W_x, Wx_bf, NPROJ * INNER);
    cvt_f32_bf16_kernel<<<blocks((long)INNER * DT_RANK, T), T, 0, stream>>>(W_dt, Wdt_bf, INNER * DT_RANK);
    cvt_f32_bf16_kernel<<<blocks((long)DIM * INNER, T), T, 0, stream>>>(W_out, Wout_bf, DIM * INNER);

    // 2. xz = x @ W_in^T
    {
        dim3 g(L_SEQ / 128, (2 * INNER) / 256);
        gemm_bf16_wmma_kernel<<<g, T, 0, stream>>>(x_bf, Win_bf, xz,
                                                   L_SEQ, 2 * INNER, DIM, nullptr, 0);
    }
    // 3. conv + silu
    conv_silu_kernel<<<blocks((long)L_SEQ * INNER, T), T, 0, stream>>>(xz, conv_w, conv_b, xc, xc_bf);

    // 4. dbc = xc @ W_x^T   (N=160 -> one edge-guarded column tile)
    {
        dim3 g(L_SEQ / 128, (NPROJ + 255) / 256);
        gemm_bf16_wmma_kernel<<<g, T, 0, stream>>>(xc_bf, Wx_bf, dbc,
                                                   L_SEQ, NPROJ, INNER, nullptr, 0);
    }
    // 5. dt_lo -> bf16
    cvt_slice_bf16_kernel<<<blocks((long)L_SEQ * DT_RANK, T), T, 0, stream>>>(
        dbc, dtlo_bf, L_SEQ, NPROJ, DT_RANK);

    // 6. delta = softplus(dtlo @ W_dt^T + b_dt)
    {
        dim3 g(L_SEQ / 128, INNER / 256);
        gemm_bf16_wmma_kernel<<<g, T, 0, stream>>>(dtlo_bf, Wdt_bf, delta,
                                                   L_SEQ, INNER, DT_RANK, b_dt, 1);
    }
    // 7. chunked selective scan
    {
        dim3 g(INNER / T, NCHUNK);
        scan_chunk_kernel<<<g, T, 0, stream>>>(delta, xc, dbc, A_log, ysbuf, hLast, prodP);
        scan_carry_kernel<<<blocks((long)INNER * STATE, T), T, 0, stream>>>(hLast, prodP, carry);
        scan_fixup_kernel<<<g, T, 0, stream>>>(delta, dbc, A_log, carry, ysbuf);
    }
    // 8. gate
    gate_kernel<<<blocks((long)L_SEQ * INNER, T), T, 0, stream>>>(ysbuf, xc, Dvec, xz, y_bf);

    // 9. out = y @ W_out^T
    {
        dim3 g(L_SEQ / 128, DIM / 256);
        gemm_bf16_wmma_kernel<<<g, T, 0, stream>>>(y_bf, Wout_bf, out,
                                                   L_SEQ, DIM, INNER, nullptr, 0);
    }
}
